// seqMnistLSTMPT_893353198022
// MI455X (gfx1250) — compile-verified, matrix operand order checked by reading, not run on hardware
//
#include <hip/hip_runtime.h>
#include <hip/hip_bf16.h>

typedef __attribute__((ext_vector_type(16))) _Float16 v16h;
typedef __attribute__((ext_vector_type(8)))  float    v8f;
typedef __attribute__((ext_vector_type(4)))  unsigned int v4u;

union F16V { v16h h; v4u q[2]; unsigned int u[8]; };

#define B_TOT  4096
#define HID    128
#define SEQ_T  120
#define FEAT_F 4
#define G4     512           // 4*H
#define TILE_B 32
#define K1     (SEQ_T * HID) // 15360
#define N1     100

// Fragment-major LDS swizzle for a B operand element (n, k) of a 16x16x32 f16 WMMA:
//   lane = ((k>>3)&1)*16 + (n&15),  v = ((k>>4)&1)*4 + ((k&7)>>1),  half = k&1
// Each lane's 8 dwords per fragment are contiguous -> 2x ds_load_b128.

// ---------------- LSTM recurrence: WMMA f16 GEMM per timestep ----------------
__global__ __launch_bounds__(256) void lstm_kernel(
    const float* __restrict__ x, const float* __restrict__ h0,
    const float* __restrict__ c0, const float* __restrict__ W_ih,
    const float* __restrict__ W_hh, const float* __restrict__ b_ih,
    const float* __restrict__ b_hh, _Float16* __restrict__ hs /*[B][T*H] f16*/)
{
  extern __shared__ char smem[];
  _Float16* whh  = (_Float16*)smem;                     // 65536 halves = 128KB, fragment-major
  _Float16* hbuf = (_Float16*)(smem + 131072);          // [32][128] halves = 8KB
  float*    wih  = (float*)(smem + 131072 + 8192);      // [512][4]  = 8KB
  float*    bias = (float*)(smem + 131072 + 8192 + 8192);        // [512] = 2KB
  float*    xs   = (float*)(smem + 131072 + 8192 + 8192 + 2048); // [32][4] = 512B

  const int tid  = threadIdx.x;
  const int row0 = blockIdx.x * TILE_B;

  // W_hh [512][128] f32 -> LDS f16, fragment-major: frag id = ks*32 + ntile
  for (int i = tid; i < G4 * HID; i += 256) {
    int n = i >> 7, k = i & 127;                         // i = n*128 + k (coalesced read)
    int ntile = n >> 4;
    int lane2 = ((k >> 3) & 1) * 16 + (n & 15);
    int v     = ((k >> 4) & 1) * 4 + ((k & 7) >> 1);
    int ks    = k >> 5;
    whh[((((ks * 32 + ntile) * 32) + lane2) * 8 + v) * 2 + (k & 1)] = (_Float16)W_hh[i];
  }
  for (int i = tid; i < G4 * FEAT_F; i += 256) wih[i] = W_ih[i];
  for (int i = tid; i < G4; i += 256)          bias[i] = b_ih[i] + b_hh[i];
  for (int i = tid; i < TILE_B * HID; i += 256) {
    int m = i >> 7, k = i & 127;
    hbuf[i] = (_Float16)h0[(size_t)(row0 + m) * HID + k];
  }

  // wave -> tile mapping: mtile = wave&1 (2 M-tiles), u0 = wave>>1 (0..3)
  // wave owns N-tiles {u0 + 4*s, s=0..7}: s = 2*q + cg -> gate q, column-group u0+4*cg
  const int wave = tid >> 5, lane = tid & 31;
  const int mtile = wave & 1, u0 = wave >> 1;
  const int hi = lane >> 4, nl = lane & 15;

  // cell state held in registers for all 120 steps
  float creg[8][2];
  #pragma unroll
  for (int r = 0; r < 8; ++r)
    #pragma unroll
    for (int cg = 0; cg < 2; ++cg) {
      int m  = mtile * 16 + r + 8 * hi;
      int jj = (u0 + 4 * cg) * 16 + nl;
      creg[r][cg] = c0[(size_t)(row0 + m) * HID + jj];
    }
  __syncthreads();

  const unsigned int* hb32 = (const unsigned int*)hbuf;
  const unsigned int* wp32 = (const unsigned int*)whh;
  const int am = mtile * 16 + nl;   // A-fragment row

  for (int t = 0; t < SEQ_T; ++t) {
    if (tid < TILE_B * FEAT_F) {
      int m = tid >> 2, f = tid & 3;
      xs[tid] = x[((size_t)(row0 + m) * SEQ_T + t) * FEAT_F + f];
    }
    __syncthreads();

    // g = h @ W_hh^T  via v_wmma_f32_16x16x32_f16
    v8f acc[8];
    #pragma unroll
    for (int s = 0; s < 8; ++s) acc[s] = (v8f){0.f,0.f,0.f,0.f,0.f,0.f,0.f,0.f};

    #pragma unroll
    for (int ks = 0; ks < 4; ++ks) {
      F16V a;
      const unsigned int* pa = hb32 + am * 64 + ks * 16 + hi * 4;
      a.q[0] = *(const v4u*)(pa);         // u[0..3]: k pairs 0..3 of this half
      a.q[1] = *(const v4u*)(pa + 8);     // u[4..7]
      #pragma unroll
      for (int s = 0; s < 8; ++s) {
        const int ntile = u0 + 4 * s;
        const unsigned int* pb = wp32 + (((ks * 32 + ntile) * 32) + lane) * 8;
        F16V b;
        b.q[0] = *(const v4u*)(pb);
        b.q[1] = *(const v4u*)(pb + 4);
        acc[s] = __builtin_amdgcn_wmma_f32_16x16x32_f16(
            false, a.h, false, b.h, (short)0, acc[s], false, false);
      }
    }
    __syncthreads();   // everyone done reading hbuf for this step

    // elementwise gates, cell update, emit h_t (registers only for gates)
    #pragma unroll
    for (int r = 0; r < 8; ++r) {
      const int m = mtile * 16 + r + 8 * hi;
      const float xf0 = xs[m * 4 + 0], xf1 = xs[m * 4 + 1];
      const float xf2 = xs[m * 4 + 2], xf3 = xs[m * 4 + 3];
      #pragma unroll
      for (int cg = 0; cg < 2; ++cg) {
        const int jj = (u0 + 4 * cg) * 16 + nl;
        float gv[4];
        #pragma unroll
        for (int q = 0; q < 4; ++q) {
          const int col = q * HID + jj;
          float gx = bias[col] + xf0 * wih[col * 4 + 0] + xf1 * wih[col * 4 + 1]
                               + xf2 * wih[col * 4 + 2] + xf3 * wih[col * 4 + 3];
          gv[q] = acc[q * 2 + cg][r] + gx;
        }
        const float ig = 1.f / (1.f + __expf(-gv[0]));
        const float fg = 1.f / (1.f + __expf(-gv[1]));
        const float gg = tanhf(gv[2]);
        const float og = 1.f / (1.f + __expf(-gv[3]));
        const float cn = fg * creg[r][cg] + ig * gg;
        creg[r][cg] = cn;
        const float hn = og * tanhf(cn);
        hbuf[m * HID + jj] = (_Float16)hn;
        hs[((size_t)(row0 + m) * SEQ_T + t) * HID + jj] = (_Float16)hn;
      }
    }
    __syncthreads();   // hbuf/xs stable before next iteration
  }
}

// ---------------- fc1: [4096,15360] x [15360,100] + ReLU, WMMA f16 ----------------
__global__ __launch_bounds__(256) void fc1_kernel(
    const _Float16* __restrict__ feat, const float* __restrict__ w,
    const float* __restrict__ bvec, float* __restrict__ hid)
{
  __shared__ _Float16 fS[64 * 32];    // [m][32] row-major (pairs contiguous)
  __shared__ _Float16 wS[32 * 128];   // fragment-major: [ntile][lane][v] dwords

  const int tid  = threadIdx.x;
  const int row0 = blockIdx.x * 64;
  const int wave = tid >> 5, lane = tid & 31;
  const int hi = lane >> 4, nl = lane & 15;
  const int mtile = wave & 3, ng = wave >> 2;   // wave: 1 M-tile x 4 N-tiles

  v8f acc[4];
  #pragma unroll
  for (int j = 0; j < 4; ++j) acc[j] = (v8f){0.f,0.f,0.f,0.f,0.f,0.f,0.f,0.f};

  const unsigned int* fg32 = (const unsigned int*)feat;  // row stride 7680 uints

  for (int k0 = 0; k0 < K1; k0 += 32) {
    for (int i = tid; i < 64 * 16; i += 256) {            // 1024 uints of feat
      int m = i >> 4, kk2 = i & 15;
      ((unsigned int*)fS)[m * 16 + kk2] =
          fg32[(size_t)(row0 + m) * (K1 / 2) + (k0 >> 1) + kk2];
    }
    for (int i = tid; i < 32 * 128; i += 256) {           // weights, zero-pad n>=100
      int n = i >> 5, kk = i & 31;
      float wv = (n < N1) ? w[(size_t)n * K1 + k0 + kk] : 0.f;
      int ntile = n >> 4;
      int lane2 = ((kk >> 3) & 1) * 16 + (n & 15);
      int v     = ((kk >> 4) & 1) * 4 + ((kk & 7) >> 1);
      wS[(((ntile * 32) + lane2) * 8 + v) * 2 + (kk & 1)] = (_Float16)wv;
    }
    if (k0 + 32 < K1) {                                   // hint next chunk
      __builtin_prefetch(&w[(size_t)(tid & 127) * K1 + k0 + 32]);
      __builtin_prefetch(&fg32[(size_t)(row0 + (tid & 63)) * (K1 / 2) + ((k0 + 32) >> 1)]);
    }
    __syncthreads();

    F16V a;
    const unsigned int* pa = (const unsigned int*)fS + (mtile * 16 + nl) * 16 + hi * 4;
    a.q[0] = *(const v4u*)(pa);
    a.q[1] = *(const v4u*)(pa + 8);
    #pragma unroll
    for (int j = 0; j < 4; ++j) {
      const int ntile = ng * 4 + j;
      const unsigned int* pb = (const unsigned int*)wS + ((ntile * 32) + lane) * 8;
      F16V b;
      b.q[0] = *(const v4u*)(pb);
      b.q[1] = *(const v4u*)(pb + 4);
      acc[j] = __builtin_amdgcn_wmma_f32_16x16x32_f16(
          false, a.h, false, b.h, (short)0, acc[j], false, false);
    }
    __syncthreads();
  }

  #pragma unroll
  for (int j = 0; j < 4; ++j) {
    const int n = (ng * 4 + j) * 16 + nl;
    if (n < N1) {
      #pragma unroll
      for (int r = 0; r < 8; ++r) {
        const int m = row0 + mtile * 16 + r + 8 * hi;
        float v = acc[j][r] + bvec[n];
        hid[(size_t)m * N1 + n] = v > 0.f ? v : 0.f;
      }
    }
  }
}

// ---------------- fc2 + log_softmax: one thread per batch row ----------------
__global__ __launch_bounds__(256) void fc2_kernel(
    const float* __restrict__ hid, const float* __restrict__ w2,
    const float* __restrict__ b2, float* __restrict__ out)
{
  __shared__ float w2s[10 * 100];
  __shared__ float b2s[10];
  const int tid = threadIdx.x;
  for (int i = tid; i < 1000; i += 256) w2s[i] = w2[i];
  if (tid < 10) b2s[tid] = b2[tid];
  __syncthreads();

  const int row = blockIdx.x * 256 + tid;
  const float* hr = hid + (size_t)row * N1;
  float logit[10];
  #pragma unroll
  for (int j = 0; j < 10; ++j) {
    float s = b2s[j];
    for (int k = 0; k < N1; ++k) s += hr[k] * w2s[j * N1 + k];
    logit[j] = s;
  }
  float mx = logit[0];
  #pragma unroll
  for (int j = 1; j < 10; ++j) mx = fmaxf(mx, logit[j]);
  float se = 0.f;
  #pragma unroll
  for (int j = 0; j < 10; ++j) se += __expf(logit[j] - mx);
  const float lse = mx + __logf(se);
  #pragma unroll
  for (int j = 0; j < 10; ++j) out[(size_t)row * 10 + j] = logit[j] - lse;
}

extern "C" void kernel_launch(void* const* d_in, const int* in_sizes, int n_in,
                              void* d_out, int out_size, void* d_ws, size_t ws_size,
                              hipStream_t stream) {
  const float* x     = (const float*)d_in[0];
  const float* h0    = (const float*)d_in[1];
  const float* c0    = (const float*)d_in[2];
  const float* W_ih  = (const float*)d_in[3];
  const float* W_hh  = (const float*)d_in[4];
  const float* b_ih  = (const float*)d_in[5];
  const float* b_hh  = (const float*)d_in[6];
  const float* fc1_w = (const float*)d_in[7];
  const float* fc1_b = (const float*)d_in[8];
  const float* fc2_w = (const float*)d_in[9];
  const float* fc2_b = (const float*)d_in[10];

  _Float16* feat = (_Float16*)d_ws;                                // [4096][15360] f16
  float*    hid  = (float*)((char*)d_ws + (size_t)B_TOT * K1 * 2); // [4096][100] f32

  const size_t lstm_smem = 131072 + 8192 + 8192 + 2048 + 512;      // 150016 B

  lstm_kernel<<<dim3(B_TOT / TILE_B), dim3(256), lstm_smem, stream>>>(
      x, h0, c0, W_ih, W_hh, b_ih, b_hh, feat);
  fc1_kernel<<<dim3(B_TOT / 64), dim3(256), 0, stream>>>(feat, fc1_w, fc1_b, hid);
  fc2_kernel<<<dim3(B_TOT / 256), dim3(256), 0, stream>>>(hid, fc2_w, fc2_b,
                                                          (float*)d_out);
}
